// PN_Actor_72997264163351
// MI455X (gfx1250) — matrix operand order repaired and checked by reading, not compiled, not run on hardware
//
#include <hip/hip_runtime.h>
#include <cstdint>
#include <cstddef>

// ---------------- problem constants (match reference) ----------------
#define BB_ 512     // batch
#define PP_ 200     // sequence / points
#define HH_ 256     // H1 == H2
#define G4_ 1024    // 4*H
#define TILE_B 16   // batches per workgroup
#define NWG (BB_ / TILE_B)   // 32 persistent workgroups

typedef __bf16 bf16_t;
typedef __attribute__((ext_vector_type(16))) __bf16 v16bf;
typedef __attribute__((ext_vector_type(8)))  float  v8f;

// ---------------- small helpers ----------------
__device__ __host__ inline unsigned short f2bf_bits(float f) {
  union { float f; unsigned u; } a; a.f = f;
  unsigned r = a.u + 0x7fffu + ((a.u >> 16) & 1u);   // round-to-nearest-even
  return (unsigned short)(r >> 16);
}
__device__ inline bf16_t f2bf(float f) {
  unsigned short h = f2bf_bits(f);
  return __builtin_bit_cast(bf16_t, h);
}
__device__ inline float bf2f(bf16_t b) {
  unsigned short h = __builtin_bit_cast(unsigned short, b);
  union { unsigned u; float f; } a; a.u = ((unsigned)h) << 16;
  return a.f;
}
__device__ inline float sigmoidf(float x) { return 1.0f / (1.0f + expf(-x)); }

// ---------------- WMMA operand loaders (per CDNA5 ISA §7.12.2) ----------------
// A: 16x32 bf16 tile, row-major source with leading dim ldk (elements).
// lanes 0-15: M=lane, K={kb..kb+7, kb+16..kb+23}; lanes 16-31: M=lane-16, K offset +8.
__device__ inline v16bf loadA(const bf16_t* base, size_t ldk, int kb) {
  const int lane = threadIdx.x & 31;
  const int m = lane & 15;
  const int koff = (lane >> 4) * 8;
  union { v16bf v; uint4 q[2]; } u;
  const bf16_t* p = base + (size_t)m * ldk + (size_t)(kb + koff);
  u.q[0] = *(const uint4*)p;
  u.q[1] = *(const uint4*)(p + 16);
  return u.v;
}
// B: 32x16 bf16 tile (K x N).  Source is W^T stored row-major [N][K], tile base
// points at row n=0 of the 16-row tile, leading dim ldk.
// lanes 0-15: N=lane, K={kb..kb+7, kb+16..kb+23}; lanes 16-31: N=lane-16, K offset +8.
__device__ inline v16bf loadB(const bf16_t* tile, size_t ldk, int kb) {
  const int lane = threadIdx.x & 31;
  const int n = lane & 15;
  const int koff = (lane >> 4) * 8;
  union { v16bf v; uint4 q[2]; } u;
  const bf16_t* p = tile + (size_t)n * ldk + (size_t)(kb + koff);
  u.q[0] = *(const uint4*)p;
  u.q[1] = *(const uint4*)(p + 16);
  return u.v;
}

// ---------------- fused LSTM gate GEMM: gates[16][1024] = xA@Wih^T + hA@Whh^T + bias ----
// 8 waves, each wave owns 8 N-tiles (64 tiles of 16 cols = 1024 gate cols).
__device__ inline void lstm_gates(const bf16_t* xA, size_t xld,
                                  const bf16_t* hA,            // LDS, ld = 256
                                  const bf16_t* __restrict__ Wih,
                                  const bf16_t* __restrict__ Whh,
                                  const float* __restrict__ bias,
                                  float* gates /* LDS 16x1024 */) {
  const int w = threadIdx.x >> 5;
  const int lane = threadIdx.x & 31;
  const int n = lane & 15, hi = lane >> 4;
  for (int i = 0; i < 8; ++i) {
    const int nt = w * 8 + i;
    const bf16_t* bih = Wih + (size_t)nt * 16 * HH_;
    const bf16_t* bhh = Whh + (size_t)nt * 16 * HH_;
    v8f acc = {};
#pragma unroll
    for (int kt = 0; kt < 8; ++kt)
      acc = __builtin_amdgcn_wmma_f32_16x16x32_bf16(
          false, loadA(xA, xld, kt * 32), false, loadB(bih, HH_, kt * 32),
          (short)0, acc, false, false);
#pragma unroll
    for (int kt = 0; kt < 8; ++kt)
      acc = __builtin_amdgcn_wmma_f32_16x16x32_bf16(
          false, loadA(hA, HH_, kt * 32), false, loadB(bhh, HH_, kt * 32),
          (short)0, acc, false, false);
    const float bv = bias[nt * 16 + n];
#pragma unroll
    for (int r = 0; r < 8; ++r)
      gates[(size_t)(r + hi * 8) * G4_ + nt * 16 + n] = acc[r] + bv;
  }
}

// q[16][256] = A[16][256] @ W  (W supplied transposed as WT[N][K]); 8 waves x 2 N-tiles.
__device__ inline void project16(const bf16_t* A /*LDS 16x256*/,
                                 const bf16_t* __restrict__ WT,
                                 float* out /* LDS 16x256 */) {
  const int w = threadIdx.x >> 5;
  const int lane = threadIdx.x & 31;
  const int n = lane & 15, hi = lane >> 4;
  for (int i = 0; i < 2; ++i) {
    const int nt = w * 2 + i;
    const bf16_t* bt = WT + (size_t)nt * 16 * HH_;
    v8f acc = {};
#pragma unroll
    for (int kt = 0; kt < 8; ++kt)
      acc = __builtin_amdgcn_wmma_f32_16x16x32_bf16(
          false, loadA(A, HH_, kt * 32), false, loadB(bt, HH_, kt * 32),
          (short)0, acc, false, false);
#pragma unroll
    for (int r = 0; r < 8; ++r)
      out[(size_t)(r + hi * 8) * HH_ + nt * 16 + n] = acc[r];
  }
}

// u[b][p] = tanh(wenc[b][p][:] + q[b][:]) . v  — one wave per row, 400 rows/wave.
__device__ inline void attn_scores(const bf16_t* __restrict__ wenc_b0,
                                   const float* qf /*LDS 16x256*/,
                                   const float* __restrict__ vvec,
                                   float* score /* LDS 16x200 */) {
  const int w = threadIdx.x >> 5;
  const int lane = threadIdx.x & 31;
  for (int r = w; r < TILE_B * PP_; r += 8) {
    const int b = r / PP_;
    const int p = r - b * PP_;
    const bf16_t* row = wenc_b0 + ((size_t)b * PP_ + p) * HH_;
    union { uint4 q; bf16_t e[8]; } u;
    u.q = *(const uint4*)(row + lane * 8);
    float s = 0.0f;
#pragma unroll
    for (int j = 0; j < 8; ++j) {
      const float tv = tanhf(bf2f(u.e[j]) + qf[b * HH_ + lane * 8 + j]);
      s += tv * vvec[lane * 8 + j];
    }
    for (int o = 16; o; o >>= 1) s += __shfl_xor(s, o);
    if (lane == 0) score[b * PP_ + p] = s;
  }
}

// ---------------- prep kernels ----------------
__global__ void cvt_bf16_kernel(const float* __restrict__ s, bf16_t* __restrict__ d, int n) {
  int i = blockIdx.x * 256 + threadIdx.x;
  if (i < n) d[i] = f2bf(s[i]);
}
__global__ void transpose256_kernel(const float* __restrict__ s, bf16_t* __restrict__ d) {
  int i = blockIdx.x * 256 + threadIdx.x;   // i = n*256 + k
  int n = i >> 8, k = i & 255;
  d[i] = f2bf(s[k * 256 + n]);
}
__global__ void addbias_kernel(const float* __restrict__ a, const float* __restrict__ b,
                               float* __restrict__ d, int n) {
  int i = blockIdx.x * 256 + threadIdx.x;
  if (i < n) d[i] = a[i] + b[i];
}
__global__ void embed_kernel(const float* __restrict__ x, const float* __restrict__ W,
                             const float* __restrict__ bias, bf16_t* __restrict__ out) {
  int i = blockIdx.x * 256 + threadIdx.x;     // over B*P*H
  if (i >= BB_ * PP_ * HH_) return;
  int h = i & 255;
  size_t bp = (size_t)(i >> 8);
  float v = fmaf(x[bp * 2 + 1], W[h * 2 + 1], fmaf(x[bp * 2], W[h * 2], bias[h]));
  out[i] = f2bf(v);
}

// ---------------- encoder: persistent, 16 batches per WG ----------------
__global__ void __launch_bounds__(256) encoder_kernel(
    const float* __restrict__ h0, const float* __restrict__ c0,
    const bf16_t* __restrict__ embed, const bf16_t* __restrict__ Wih,
    const bf16_t* __restrict__ Whh, const float* __restrict__ bias,
    bf16_t* __restrict__ enc, float* __restrict__ henc) {
  extern __shared__ char smem[];
  float* gates = (float*)smem;                       // 16*1024 f32
  float* cst = gates + TILE_B * G4_;                 // 16*256 f32
  bf16_t* hbf = (bf16_t*)(cst + TILE_B * HH_);       // 16*256 bf16
  const int tid = threadIdx.x;                       // hidden column
  const int batch0 = blockIdx.x * TILE_B;

  for (int b = 0; b < TILE_B; ++b) {
    hbf[b * HH_ + tid] = f2bf(h0[(size_t)(batch0 + b) * HH_ + tid]);
    cst[b * HH_ + tid] = c0[(size_t)(batch0 + b) * HH_ + tid];
  }
  __syncthreads();

  for (int t = 0; t < PP_; ++t) {
    lstm_gates(embed + ((size_t)batch0 * PP_ + t) * HH_, (size_t)PP_ * HH_,
               hbf, Wih, Whh, bias, gates);
    __syncthreads();
    for (int b = 0; b < TILE_B; ++b) {
      const float gi = gates[b * G4_ + tid];
      const float gf = gates[b * G4_ + 256 + tid];
      const float gg = gates[b * G4_ + 512 + tid];
      const float go = gates[b * G4_ + 768 + tid];
      float c = sigmoidf(gf) * cst[b * HH_ + tid] + sigmoidf(gi) * tanhf(gg);
      float h = sigmoidf(go) * tanhf(c);
      cst[b * HH_ + tid] = c;
      hbf[b * HH_ + tid] = f2bf(h);
      enc[(((size_t)(batch0 + b)) * PP_ + t) * HH_ + tid] = f2bf(h);
      if (t == PP_ - 1) henc[(size_t)(batch0 + b) * HH_ + tid] = h;
    }
    __syncthreads();
  }
}

// ---------------- wenc / wenc_g projection: [B*P,256] @ W^T[N][K] ----------------
__global__ void __launch_bounds__(256) proj_kernel(const bf16_t* __restrict__ A,
                                                   const bf16_t* __restrict__ WT,
                                                   bf16_t* __restrict__ out) {
  const int w = threadIdx.x >> 5;
  const int lane = threadIdx.x & 31;
  const int n = lane & 15, hi = lane >> 4;
  const bf16_t* Ab = A + (size_t)blockIdx.x * 16 * HH_;
  bf16_t* Ob = out + (size_t)blockIdx.x * 16 * HH_;
  for (int i = 0; i < 2; ++i) {
    const int nt = w * 2 + i;
    const bf16_t* bt = WT + (size_t)nt * 16 * HH_;
    v8f acc = {};
#pragma unroll
    for (int kt = 0; kt < 8; ++kt)
      acc = __builtin_amdgcn_wmma_f32_16x16x32_bf16(
          false, loadA(Ab, HH_, kt * 32), false, loadB(bt, HH_, kt * 32),
          (short)0, acc, false, false);
#pragma unroll
    for (int r = 0; r < 8; ++r)
      Ob[(size_t)(r + hi * 8) * HH_ + nt * 16 + n] = f2bf(acc[r]);
  }
}

// ---------------- decoder: persistent glimpse-pointer scan ----------------
__global__ void __launch_bounds__(256) decoder_kernel(
    const float* __restrict__ henc, const float* __restrict__ c0d,
    const float* __restrict__ dec0, const bf16_t* __restrict__ embed,
    const bf16_t* __restrict__ enc, const bf16_t* __restrict__ wencg,
    const bf16_t* __restrict__ wencp, const bf16_t* __restrict__ Wih,
    const bf16_t* __restrict__ Whh, const float* __restrict__ bias,
    const bf16_t* __restrict__ WqTg, const bf16_t* __restrict__ WqTp,
    const float* __restrict__ vg, const float* __restrict__ vp,
    float* __restrict__ out) {
  extern __shared__ char smem[];
  float* gates = (float*)smem;                         // 16*1024
  float* cst    = gates + TILE_B * G4_;                // 16*256
  float* qf     = cst + TILE_B * HH_;                  // 16*256
  float* score  = qf + TILE_B * HH_;                   // 16*200
  float* maskf  = score + TILE_B * PP_;                // 16*200
  float* aw     = maskf + TILE_B * PP_;                // 16*200
  float* logacc = aw + TILE_B * PP_;                   // 16
  int*   redI   = (int*)(logacc + TILE_B);             // 16
  bf16_t* hbf   = (bf16_t*)(redI + TILE_B);            // 16*256
  bf16_t* xtbf  = hbf + TILE_B * HH_;                  // 16*256
  bf16_t* gbf   = xtbf + TILE_B * HH_;                 // 16*256

  const int tid = threadIdx.x;
  const int w = tid >> 5;
  const int lane = tid & 31;
  const int batch0 = blockIdx.x * TILE_B;
  const bf16_t* wencg_b0 = wencg + (size_t)batch0 * PP_ * HH_;
  const bf16_t* wencp_b0 = wencp + (size_t)batch0 * PP_ * HH_;

  for (int b = 0; b < TILE_B; ++b) {
    hbf[b * HH_ + tid]  = f2bf(henc[(size_t)(batch0 + b) * HH_ + tid]);
    cst[b * HH_ + tid]  = c0d[(size_t)(batch0 + b) * HH_ + tid];
    xtbf[b * HH_ + tid] = f2bf(dec0[(size_t)(batch0 + b) * HH_ + tid]);
  }
  for (int i = tid; i < TILE_B * PP_; i += 256) maskf[i] = 0.0f;
  if (tid < TILE_B) logacc[tid] = 0.0f;
  __syncthreads();

  for (int t = 0; t < PP_; ++t) {
    // --- decoder LSTM cell ---
    lstm_gates(xtbf, (size_t)HH_, hbf, Wih, Whh, bias, gates);
    __syncthreads();
    for (int b = 0; b < TILE_B; ++b) {
      const float gi = gates[b * G4_ + tid];
      const float gf = gates[b * G4_ + 256 + tid];
      const float gg = gates[b * G4_ + 512 + tid];
      const float go = gates[b * G4_ + 768 + tid];
      float c = sigmoidf(gf) * cst[b * HH_ + tid] + sigmoidf(gi) * tanhf(gg);
      float h = sigmoidf(go) * tanhf(c);
      cst[b * HH_ + tid] = c;
      hbf[b * HH_ + tid] = f2bf(h);
    }
    __syncthreads();
    // --- glimpse: q = h @ Wq_g ---
    project16(hbf, WqTg, qf);
    __syncthreads();
    attn_scores(wencg_b0, qf, vg, score);
    __syncthreads();
    // --- glimpse softmax (mask applied) ---
    for (int bb = w; bb < TILE_B; bb += 8) {
      float mx = -3.0e38f;
      for (int p = lane; p < PP_; p += 32) {
        float l = 10.0f * tanhf(score[bb * PP_ + p]) - maskf[bb * PP_ + p] * 1.0e8f;
        score[bb * PP_ + p] = l;
        mx = fmaxf(mx, l);
      }
      for (int o = 16; o; o >>= 1) mx = fmaxf(mx, __shfl_xor(mx, o));
      float s = 0.0f;
      for (int p = lane; p < PP_; p += 32) s += expf(score[bb * PP_ + p] - mx);
      for (int o = 16; o; o >>= 1) s += __shfl_xor(s, o);
      const float inv = 1.0f / s;
      for (int p = lane; p < PP_; p += 32)
        aw[bb * PP_ + p] = expf(score[bb * PP_ + p] - mx) * inv;
    }
    __syncthreads();
    // --- g = einsum(enc, a) + h ---
    for (int b = 0; b < TILE_B; ++b) {
      float acc = bf2f(hbf[b * HH_ + tid]);
      const bf16_t* erow = enc + ((size_t)(batch0 + b) * PP_) * HH_ + tid;
      const float* arow = aw + b * PP_;
      for (int p = 0; p < PP_; ++p) acc += arow[p] * bf2f(erow[(size_t)p * HH_]);
      gbf[b * HH_ + tid] = f2bf(acc);
    }
    __syncthreads();
    // --- pointer: q = g @ Wq ---
    project16(gbf, WqTp, qf);
    __syncthreads();
    attn_scores(wencp_b0, qf, vp, score);
    __syncthreads();
    // --- pointer softmax + greedy argmax; prob(argmax) = 1/sumexp ---
    for (int bb = w; bb < TILE_B; bb += 8) {
      float mx = -3.0e38f; int mi = 0;
      for (int p = lane; p < PP_; p += 32) {
        float l = 10.0f * tanhf(score[bb * PP_ + p]) - maskf[bb * PP_ + p] * 1.0e8f;
        score[bb * PP_ + p] = l;
        if (l > mx) { mx = l; mi = p; }
      }
      for (int o = 16; o; o >>= 1) {
        float om = __shfl_xor(mx, o); int oi = __shfl_xor(mi, o);
        if (om > mx || (om == mx && oi < mi)) { mx = om; mi = oi; }
      }
      float s = 0.0f;
      for (int p = lane; p < PP_; p += 32) s += expf(score[bb * PP_ + p] - mx);
      for (int o = 16; o; o >>= 1) s += __shfl_xor(s, o);
      if (lane == 0) {
        redI[bb] = mi;
        logacc[bb] += -logf(s);
        out[(size_t)(batch0 + bb) * (PP_ + 1) + t] = (float)mi;
        if (t == 0) out[(size_t)(batch0 + bb) * (PP_ + 1) + PP_] = (float)mi;
      }
    }
    __syncthreads();
    // --- feed back chosen embedding; update mask ---
    for (int b = 0; b < TILE_B; ++b)
      xtbf[b * HH_ + tid] =
          embed[(((size_t)(batch0 + b)) * PP_ + redI[b]) * HH_ + tid];
    if (tid < TILE_B) maskf[tid * PP_ + redI[tid]] += 1.0f;
    __syncthreads();
  }
  if (tid < TILE_B) out[(size_t)BB_ * (PP_ + 1) + batch0 + tid] = logacc[tid];
}

// ---------------- host launcher ----------------
extern "C" void kernel_launch(void* const* d_in, const int* in_sizes, int n_in,
                              void* d_out, int out_size, void* d_ws, size_t ws_size,
                              hipStream_t stream) {
  (void)in_sizes; (void)n_in; (void)out_size; (void)ws_size;
  const float* x        = (const float*)d_in[0];
  const float* W_emb    = (const float*)d_in[1];
  const float* b_emb    = (const float*)d_in[2];
  const float* W_ih_enc = (const float*)d_in[3];
  const float* W_hh_enc = (const float*)d_in[4];
  const float* b_ih_enc = (const float*)d_in[5];
  const float* b_hh_enc = (const float*)d_in[6];
  const float* h0_enc   = (const float*)d_in[7];
  const float* c0_enc   = (const float*)d_in[8];
  const float* W_ih_dec = (const float*)d_in[9];
  const float* W_hh_dec = (const float*)d_in[10];
  const float* b_ih_dec = (const float*)d_in[11];
  const float* b_hh_dec = (const float*)d_in[12];
  const float* dec0     = (const float*)d_in[13];
  const float* c0_dec   = (const float*)d_in[14];
  const float* Wref_g   = (const float*)d_in[15];
  const float* Wq_g     = (const float*)d_in[16];
  const float* v_g      = (const float*)d_in[17];
  const float* Wref     = (const float*)d_in[18];
  const float* Wq       = (const float*)d_in[19];
  const float* v        = (const float*)d_in[20];

  char* ws = (char*)d_ws;
  size_t off = 0;
  auto alloc = [&](size_t bytes) -> char* {
    char* p = ws + off;
    off = (off + bytes + 255) & ~(size_t)255;
    return p;
  };
  const size_t actN = (size_t)BB_ * PP_ * HH_;
  bf16_t* embed_bf = (bf16_t*)alloc(actN * 2);
  bf16_t* enc_bf   = (bf16_t*)alloc(actN * 2);
  bf16_t* wenc_bf  = (bf16_t*)alloc(actN * 2);
  bf16_t* wencg_bf = (bf16_t*)alloc(actN * 2);
  bf16_t* Wih_e = (bf16_t*)alloc((size_t)G4_ * HH_ * 2);
  bf16_t* Whh_e = (bf16_t*)alloc((size_t)G4_ * HH_ * 2);
  bf16_t* Wih_d = (bf16_t*)alloc((size_t)G4_ * HH_ * 2);
  bf16_t* Whh_d = (bf16_t*)alloc((size_t)G4_ * HH_ * 2);
  bf16_t* WrefTg = (bf16_t*)alloc((size_t)HH_ * HH_ * 2);
  bf16_t* WqTg   = (bf16_t*)alloc((size_t)HH_ * HH_ * 2);
  bf16_t* WrefT  = (bf16_t*)alloc((size_t)HH_ * HH_ * 2);
  bf16_t* WqT    = (bf16_t*)alloc((size_t)HH_ * HH_ * 2);
  float* benc = (float*)alloc((size_t)G4_ * 4);
  float* bdec = (float*)alloc((size_t)G4_ * 4);
  float* henc = (float*)alloc((size_t)BB_ * HH_ * 4);

  const int nW = G4_ * HH_;
  cvt_bf16_kernel<<<(nW + 255) / 256, 256, 0, stream>>>(W_ih_enc, Wih_e, nW);
  cvt_bf16_kernel<<<(nW + 255) / 256, 256, 0, stream>>>(W_hh_enc, Whh_e, nW);
  cvt_bf16_kernel<<<(nW + 255) / 256, 256, 0, stream>>>(W_ih_dec, Wih_d, nW);
  cvt_bf16_kernel<<<(nW + 255) / 256, 256, 0, stream>>>(W_hh_dec, Whh_d, nW);
  transpose256_kernel<<<256, 256, 0, stream>>>(Wref_g, WrefTg);
  transpose256_kernel<<<256, 256, 0, stream>>>(Wq_g,   WqTg);
  transpose256_kernel<<<256, 256, 0, stream>>>(Wref,   WrefT);
  transpose256_kernel<<<256, 256, 0, stream>>>(Wq,     WqT);
  addbias_kernel<<<4, 256, 0, stream>>>(b_ih_enc, b_hh_enc, benc, G4_);
  addbias_kernel<<<4, 256, 0, stream>>>(b_ih_dec, b_hh_dec, bdec, G4_);
  embed_kernel<<<(int)((actN + 255) / 256), 256, 0, stream>>>(x, W_emb, b_emb, embed_bf);

  const size_t encSmem = (size_t)TILE_B * G4_ * 4 + (size_t)TILE_B * HH_ * 4 +
                         (size_t)TILE_B * HH_ * 2;                       // 90112 B
  encoder_kernel<<<NWG, 256, encSmem, stream>>>(h0_enc, c0_enc, embed_bf,
                                                Wih_e, Whh_e, benc, enc_bf, henc);

  proj_kernel<<<(BB_ * PP_) / 16, 256, 0, stream>>>(enc_bf, WrefTg, wencg_bf);
  proj_kernel<<<(BB_ * PP_) / 16, 256, 0, stream>>>(enc_bf, WrefT,  wenc_bf);

  const size_t decSmem = (size_t)TILE_B * G4_ * 4      // gates
                       + (size_t)TILE_B * HH_ * 4 * 2  // cst, qf
                       + (size_t)TILE_B * PP_ * 4 * 3  // score, mask, aw
                       + (size_t)TILE_B * 4 * 2        // logacc, redI
                       + (size_t)TILE_B * HH_ * 2 * 3; // hbf, xtbf, gbf  => 161408 B
  decoder_kernel<<<NWG, 256, decSmem, stream>>>(
      henc, c0_dec, dec0, embed_bf, enc_bf, wencg_bf, wenc_bf,
      Wih_d, Whh_d, bdec, WqTg, WqT, v_g, v, (float*)d_out);
}
